// SimpleLSTM_62861141344719
// MI455X (gfx1250) — matrix operand order repaired and checked by reading, not compiled, not run on hardware
//
#include <hip/hip_runtime.h>
#include <hip/hip_bf16.h>

#define VOCAB  32000
#define VPAD   32256          // 63 * 512, zero-padded FC weight rows
#define EMBED  512
#define HIDDEN 1024
#define GATES  (4 * HIDDEN)   // 4096
#define BB     64
#define TT     512

typedef __bf16 bf16;
typedef __attribute__((ext_vector_type(16))) __bf16 v16bf;
typedef __attribute__((ext_vector_type(8)))  __bf16 v8bf;
typedef __attribute__((ext_vector_type(8)))  float  v8f;

__device__ __forceinline__ v16bf mk_a(v8bf lo, v8bf hi) {
  v16bf a;
#pragma unroll
  for (int i = 0; i < 8; ++i) { a[i] = lo[i]; a[i + 8] = hi[i]; }
  return a;
}

// ---------------------------------------------------------------------------
// Shared GEMM core, software-pipelined (double-buffered fragments).
// One block = 256 threads = 8 waves, one 16-row A tile staged in LDS, each
// wave computes a 16x64 output strip (4 accumulators, A fragment reused 4x).
// B is [N,K] row-major bf16 (natural W storage for x @ W^T): lane L holds
// column L&15, K = (L>>4)*16 + [0..15] contiguous (one 32B load).
// A fragment (from LDS): lane L holds row L&15, K = (L>>4)*16 + [0..15] contiguous.
// ---------------------------------------------------------------------------
template <int K>
__device__ __forceinline__ void gemm_core(const bf16* As,              // LDS [16][K]
                                          const bf16* __restrict__ Bb, // Wt + n0w*K
                                          v8f acc[4], int lane) {
  int row  = lane & 15;
  int koff = (lane >> 4) << 4;   // 0 or 16
  int kblk = (lane >> 4) << 4;   // 0 or 16
  int n    = lane & 15;
  const bf16* Arow = As + row * K + koff;
  const bf16* Bp0  = Bb + (size_t)(n)      * K + kblk;
  const bf16* Bp1  = Bb + (size_t)(16 + n) * K + kblk;
  const bf16* Bp2  = Bb + (size_t)(32 + n) * K + kblk;
  const bf16* Bp3  = Bb + (size_t)(48 + n) * K + kblk;

  // Prologue: fetch k=0 fragments.
  v8bf  lo = *(const v8bf*)(Arow);
  v8bf  hi = *(const v8bf*)(Arow + 8);
  v16bf b0 = *(const v16bf*)(Bp0);
  v16bf b1 = *(const v16bf*)(Bp1);
  v16bf b2 = *(const v16bf*)(Bp2);
  v16bf b3 = *(const v16bf*)(Bp3);

  // Steady state: issue loads for k, then WMMAs for k-32.
#pragma unroll 2
  for (int k = 32; k < K; k += 32) {
    v8bf  lo_n = *(const v8bf*)(Arow + k);
    v8bf  hi_n = *(const v8bf*)(Arow + k + 8);
    v16bf c0 = *(const v16bf*)(Bp0 + k);
    v16bf c1 = *(const v16bf*)(Bp1 + k);
    v16bf c2 = *(const v16bf*)(Bp2 + k);
    v16bf c3 = *(const v16bf*)(Bp3 + k);
    v16bf a = mk_a(lo, hi);
    acc[0] = __builtin_amdgcn_wmma_f32_16x16x32_bf16(false, a, false, b0, (short)0,
                                                     acc[0], false, false);
    acc[1] = __builtin_amdgcn_wmma_f32_16x16x32_bf16(false, a, false, b1, (short)0,
                                                     acc[1], false, false);
    acc[2] = __builtin_amdgcn_wmma_f32_16x16x32_bf16(false, a, false, b2, (short)0,
                                                     acc[2], false, false);
    acc[3] = __builtin_amdgcn_wmma_f32_16x16x32_bf16(false, a, false, b3, (short)0,
                                                     acc[3], false, false);
    lo = lo_n; hi = hi_n; b0 = c0; b1 = c1; b2 = c2; b3 = c3;
  }

  // Epilogue: last K-slice.
  {
    v16bf a = mk_a(lo, hi);
    acc[0] = __builtin_amdgcn_wmma_f32_16x16x32_bf16(false, a, false, b0, (short)0,
                                                     acc[0], false, false);
    acc[1] = __builtin_amdgcn_wmma_f32_16x16x32_bf16(false, a, false, b1, (short)0,
                                                     acc[1], false, false);
    acc[2] = __builtin_amdgcn_wmma_f32_16x16x32_bf16(false, a, false, b2, (short)0,
                                                     acc[2], false, false);
    acc[3] = __builtin_amdgcn_wmma_f32_16x16x32_bf16(false, a, false, b3, (short)0,
                                                     acc[3], false, false);
  }
}

// Cooperative copy of the block's 16xK A tile (contiguous, lda==K) into LDS.
template <int K>
__device__ __forceinline__ void stage_a(const bf16* __restrict__ Aglob, bf16* As) {
  const uint4* src = (const uint4*)Aglob;
  uint4* dst = (uint4*)As;
  constexpr int nchunk = 16 * K * 2 / 16;     // 16-byte chunks
#pragma unroll
  for (int i = 0; i < nchunk / 256; ++i)
    dst[i * 256 + threadIdx.x] = src[i * 256 + threadIdx.x];
  __syncthreads();
}

// ---------------------------------------------------------------------------
// Elementwise helpers
// ---------------------------------------------------------------------------

__global__ void k_cast_bf16(const float* __restrict__ in, bf16* __restrict__ out, int n) {
  int i = blockIdx.x * blockDim.x + threadIdx.x;
  if (i < n) out[i] = (bf16)in[i];
}

// Cast with zero-padding (for W_fc rows 32000 -> 32256).
__global__ void k_cast_pad(const float* __restrict__ in, bf16* __restrict__ out,
                           int n_in, int n_out) {
  int i = blockIdx.x * blockDim.x + threadIdx.x;
  if (i < n_out) out[i] = (i < n_in) ? (bf16)in[i] : (bf16)0.0f;
}

// Embedding gather, fp32 table -> bf16 activations. idx over B*T*EMBED.
__global__ void k_embed(const int* __restrict__ x, const float* __restrict__ table,
                        bf16* __restrict__ out) {
  int idx = blockIdx.x * blockDim.x + threadIdx.x;
  int bt  = idx >> 9;                 // EMBED == 512
  int e   = idx & (EMBED - 1);
  out[idx] = (bf16)table[(size_t)x[bt] * EMBED + e];
}

__global__ void k_init_state(bf16* __restrict__ h, float* __restrict__ c) {
  int i = blockIdx.x * blockDim.x + threadIdx.x;
  if (i < BB * HIDDEN) { h[i] = (bf16)0.0f; c[i] = 0.0f; }
}

// ---------------------------------------------------------------------------
// Input projection: xp[t][b][g] = emb[b*T+t] . W_ih[g] + b_ih[g] + b_hh[g]
// M = 32768, N = 4096, K = 512. grid = (2048, 8), block = 256.
// ---------------------------------------------------------------------------
__global__ void __launch_bounds__(256) k_xproj(const bf16* __restrict__ emb,
                                               const bf16* __restrict__ Wih,
                                               const float* __restrict__ b_ih,
                                               const float* __restrict__ b_hh,
                                               float* __restrict__ xp) {
  __shared__ bf16 As[16 * EMBED];
  int lane = threadIdx.x & 31;
  int wave = threadIdx.x >> 5;
  int m0  = blockIdx.x * 16;
  int n0w = (blockIdx.y * 8 + wave) * 64;
  int n   = lane & 15;

  stage_a<EMBED>(emb + (size_t)m0 * EMBED, As);

  v8f acc[4];
#pragma unroll
  for (int j = 0; j < 4; ++j) {
    float bias = b_ih[n0w + j * 16 + n] + b_hh[n0w + j * 16 + n];
#pragma unroll
    for (int r = 0; r < 8; ++r) acc[j][r] = bias;
  }

  gemm_core<EMBED>(As, Wih + (size_t)n0w * EMBED, acc, lane);

#pragma unroll
  for (int j = 0; j < 4; ++j)
#pragma unroll
    for (int r = 0; r < 8; ++r) {
      int m  = m0 + r + ((lane >> 4) << 3);   // global row bt = b*T + t
      int bb = m >> 9;                        // T == 512
      int tt = m & (TT - 1);
      xp[((size_t)tt * BB + bb) * GATES + n0w + j * 16 + n] = acc[j][r];
    }
}

// ---------------------------------------------------------------------------
// Recurrence step GEMM: g[b][n] = xp_t[b][n] + h[b] . W_hh[n]
// M = 64, N = 4096, K = 1024. grid = (4, 8), block = 256.
// ---------------------------------------------------------------------------
__global__ void __launch_bounds__(256) k_step_gemm(const bf16* __restrict__ h,
                                                   const bf16* __restrict__ Whh,
                                                   const float* __restrict__ xp_t,
                                                   float* __restrict__ g) {
  __shared__ bf16 As[16 * HIDDEN];
  int lane = threadIdx.x & 31;
  int wave = threadIdx.x >> 5;
  int m0  = blockIdx.x * 16;
  int n0w = (blockIdx.y * 8 + wave) * 64;
  int n   = lane & 15;

  stage_a<HIDDEN>(h + (size_t)m0 * HIDDEN, As);

  v8f acc[4];
#pragma unroll
  for (int j = 0; j < 4; ++j)
#pragma unroll
    for (int r = 0; r < 8; ++r) {
      int m = m0 + r + ((lane >> 4) << 3);
      acc[j][r] = xp_t[(size_t)m * GATES + n0w + j * 16 + n];
    }

  gemm_core<HIDDEN>(As, Whh + (size_t)n0w * HIDDEN, acc, lane);

#pragma unroll
  for (int j = 0; j < 4; ++j)
#pragma unroll
    for (int r = 0; r < 8; ++r) {
      int m = m0 + r + ((lane >> 4) << 3);
      g[(size_t)m * GATES + n0w + j * 16 + n] = acc[j][r];
    }
}

// ---------------------------------------------------------------------------
// Gate nonlinearities + state update. PyTorch gate order i, f, g, o.
// ---------------------------------------------------------------------------
__global__ void k_gates(const float* __restrict__ g, float* __restrict__ c,
                        bf16* __restrict__ h) {
  int idx = blockIdx.x * blockDim.x + threadIdx.x;
  if (idx >= BB * HIDDEN) return;
  int b = idx >> 10;                        // HIDDEN == 1024
  int j = idx & (HIDDEN - 1);
  const float* gb = g + (size_t)b * GATES;
  float ig = gb[j];
  float fg = gb[HIDDEN + j];
  float gg = gb[2 * HIDDEN + j];
  float og = gb[3 * HIDDEN + j];
  float si = 1.0f / (1.0f + __expf(-ig));
  float sf = 1.0f / (1.0f + __expf(-fg));
  float tg = tanhf(gg);
  float so = 1.0f / (1.0f + __expf(-og));
  float cn = sf * c[idx] + si * tg;
  c[idx] = cn;
  h[idx] = (bf16)(so * tanhf(cn));
}

// ---------------------------------------------------------------------------
// Final FC: out[b][v] = h[b] . W_fc[v] + b_fc[v]. M=64, N=32000 (padded 32256),
// K=1024. grid = (4, 63), block = 256. Guards are wave-uniform (EXEC stays 1s).
// ---------------------------------------------------------------------------
__global__ void __launch_bounds__(256) k_fc(const bf16* __restrict__ h,
                                            const bf16* __restrict__ Wfc,
                                            const float* __restrict__ b_fc,
                                            float* __restrict__ out) {
  __shared__ bf16 As[16 * HIDDEN];
  int lane = threadIdx.x & 31;
  int wave = threadIdx.x >> 5;
  int m0  = blockIdx.x * 16;
  int n0w = (blockIdx.y * 8 + wave) * 64;
  int n   = lane & 15;

  stage_a<HIDDEN>(h + (size_t)m0 * HIDDEN, As);

  v8f acc[4];
#pragma unroll
  for (int j = 0; j < 4; ++j) {
    int n0t = n0w + j * 16;
    float bias = (n0t < VOCAB) ? b_fc[n0t + n] : 0.0f;
#pragma unroll
    for (int r = 0; r < 8; ++r) acc[j][r] = bias;
  }

  gemm_core<HIDDEN>(As, Wfc + (size_t)n0w * HIDDEN, acc, lane);

#pragma unroll
  for (int j = 0; j < 4; ++j) {
    int n0t = n0w + j * 16;
    if (n0t < VOCAB) {
#pragma unroll
      for (int r = 0; r < 8; ++r) {
        int m = m0 + r + ((lane >> 4) << 3);
        out[(size_t)m * VOCAB + n0t + n] = acc[j][r];
      }
    }
  }
}

// ---------------------------------------------------------------------------
// Host-side orchestration
// ---------------------------------------------------------------------------
extern "C" void kernel_launch(void* const* d_in, const int* in_sizes, int n_in,
                              void* d_out, int out_size, void* d_ws, size_t ws_size,
                              hipStream_t stream) {
  const int*   x         = (const int*)  d_in[0];
  const float* emb_table = (const float*)d_in[1];
  const float* W_ih      = (const float*)d_in[2];
  const float* W_hh      = (const float*)d_in[3];
  const float* b_ih      = (const float*)d_in[4];
  const float* b_hh      = (const float*)d_in[5];
  const float* W_fc      = (const float*)d_in[6];
  const float* b_fc      = (const float*)d_in[7];
  float* out = (float*)d_out;

  // Workspace carve-up (256B aligned regions).
  char*  ws  = (char*)d_ws;
  size_t off = 0;
  auto carve = [&](size_t bytes) -> void* {
    void* p = ws + off;
    off += (bytes + 255) & ~(size_t)255;
    return p;
  };
  bf16*  Wih_b  = (bf16*) carve((size_t)GATES * EMBED  * sizeof(bf16));   //   4 MiB
  bf16*  Whh_b  = (bf16*) carve((size_t)GATES * HIDDEN * sizeof(bf16));   //   8 MiB
  bf16*  Wfc_b  = (bf16*) carve((size_t)VPAD  * HIDDEN * sizeof(bf16));   //  63 MiB
  bf16*  emb_b  = (bf16*) carve((size_t)BB * TT * EMBED * sizeof(bf16));  //  32 MiB
  float* xp     = (float*)carve((size_t)TT * BB * GATES * sizeof(float)); // 512 MiB
  float* gbuf   = (float*)carve((size_t)BB * GATES * sizeof(float));      //   1 MiB
  bf16*  h_st   = (bf16*) carve((size_t)BB * HIDDEN * sizeof(bf16));
  float* c_st   = (float*)carve((size_t)BB * HIDDEN * sizeof(float));

  // 1) Weight casts to bf16 ([N,K] row-major is already WMMA-B storage).
  {
    int n1 = GATES * EMBED;
    k_cast_bf16<<<(n1 + 255) / 256, 256, 0, stream>>>(W_ih, Wih_b, n1);
    int n2 = GATES * HIDDEN;
    k_cast_bf16<<<(n2 + 255) / 256, 256, 0, stream>>>(W_hh, Whh_b, n2);
    int n3i = VOCAB * HIDDEN, n3o = VPAD * HIDDEN;
    k_cast_pad<<<(n3o + 255) / 256, 256, 0, stream>>>(W_fc, Wfc_b, n3i, n3o);
  }

  // 2) Embedding gather (fp32 -> bf16).
  k_embed<<<(BB * TT * EMBED) / 256, 256, 0, stream>>>(x, emb_table, emb_b);

  // 3) Input projection GEMM with fused bias: xp[t][b][4H].
  k_xproj<<<dim3((BB * TT) / 16, GATES / 512), 256, 0, stream>>>(
      emb_b, Wih_b, b_ih, b_hh, xp);

  // 4) Zero LSTM state.
  k_init_state<<<(BB * HIDDEN + 255) / 256, 256, 0, stream>>>(h_st, c_st);

  // 5) Recurrence: 512 steps of (WMMA GEMM seeded from xp) + gate update.
  for (int t = 0; t < TT; ++t) {
    const float* xp_t = xp + (size_t)t * BB * GATES;
    k_step_gemm<<<dim3(BB / 16, GATES / 512), 256, 0, stream>>>(
        h_st, Whh_b, xp_t, gbuf);
    k_gates<<<(BB * HIDDEN) / 256, 256, 0, stream>>>(gbuf, c_st, h_st);
  }

  // 6) Final FC with fused bias -> d_out [64, 32000] fp32.
  k_fc<<<dim3(BB / 16, VPAD / 512), 256, 0, stream>>>(h_st, Wfc_b, b_fc, out);
}
